// GPTLanguageModel_78881369358386
// MI455X (gfx1250) — matrix operand order repaired and checked by reading, not compiled, not run on hardware
//
#include <hip/hip_runtime.h>

typedef __attribute__((ext_vector_type(16))) _Float16 v16h;
typedef __attribute__((ext_vector_type(8)))  float    v8f;
typedef _Float16 h16t;

// ---------------------------------------------------------------------------
// WMMA helpers (CDNA5 gfx1250, wave32). Layouts per CDNA5 ISA 7.12.2.
// ---------------------------------------------------------------------------
static __device__ __forceinline__ v8f zero8() {
  v8f z = {0.f, 0.f, 0.f, 0.f, 0.f, 0.f, 0.f, 0.f};
  return z;
}

static __device__ __forceinline__ v8f wmma32f16(v16h a, v16h b, v8f c) {
  return __builtin_amdgcn_wmma_f32_16x16x32_f16(false, a, false, b, (short)0, c,
                                                false, false);
}

// raw v_exp_f32 (exp2), no denorm-safe expansion -- fine for softmax
static __device__ __forceinline__ float exp2raw(float x) {
  return __builtin_amdgcn_exp2f(x);
}

// A-matrix fragment 16x32 f16. Storage: row-major, rows indexed by lane%16,
// stride in halves. K slots per ISA table:
//   lanes 0-15 : V0..V3 -> K pairs 0..7, V4..V7 -> K 16..23
//   lanes 16-31: V0..V3 -> K pairs 8..15, V4..V7 -> K 24..31
static __device__ __forceinline__ v16h load_a_frag(const h16t* base, int stride) {
  int lane = (int)(threadIdx.x & 31);
  const h16t* rp = base + (lane & 15) * stride;
  int kadd = (lane & 16) ? 8 : 0;
  union { v16h v; unsigned int u[8]; } r;
#pragma unroll
  for (int p = 0; p < 8; ++p) {
    int k0 = ((p < 4) ? (p * 2) : (16 + (p - 4) * 2)) + kadd;
    r.u[p] = *(const unsigned int*)(rp + k0);
  }
  return r.v;
}

// B-matrix fragment 32x16 f16 from [n][k-contiguous] storage:
// lane n = lane%16, K = (lane<16 ? 0 : 8) + slot (slots 0-7), +16 for slots 8-15.
static __device__ __forceinline__ v16h load_b_frag(const h16t* base, int stride) {
  int lane = (int)(threadIdx.x & 31);
  const h16t* p = base + (lane & 15) * stride + ((lane & 16) ? 8 : 0);
  union { v16h v; uint4 q[2]; } r;
  r.q[0] = *(const uint4*)(p);
  r.q[1] = *(const uint4*)(p + 16);
  return r.v;
}

// DPP lane-move within 16-lane row (wave32): VALU-pipe butterfly, no LDS.
template <int CTRL>
static __device__ __forceinline__ float dppf(float v) {
  int x = __builtin_amdgcn_update_dpp(0, __builtin_bit_cast(int, v), CTRL, 0xF,
                                      0xF, true);
  return __builtin_bit_cast(float, x);
}

// 16-lane butterfly reduce (valid within each 16-lane row group):
// quad_perm xor1 (0xB1), xor2 (0x4E), row_half_mirror (0x141), row_mirror (0x140)
static __device__ __forceinline__ float red16_max(float v) {
  float t;
  t = dppf<0xB1>(v);  v = (t > v) ? t : v;
  t = dppf<0x4E>(v);  v = (t > v) ? t : v;
  t = dppf<0x141>(v); v = (t > v) ? t : v;
  t = dppf<0x140>(v); v = (t > v) ? t : v;
  return v;
}
static __device__ __forceinline__ float red16_sum(float v) {
  v += dppf<0xB1>(v);
  v += dppf<0x4E>(v);
  v += dppf<0x141>(v);
  v += dppf<0x140>(v);
  return v;
}

// ---------------------------------------------------------------------------
// Model constants
// ---------------------------------------------------------------------------
#define NV 65
#define ND 384
#define NH 6
#define NHD 64
#define NL 6
#define NT 256
#define NB 64
#define BT (NB * NT)          // 16384
#define DFF (4 * ND)          // 1536
#define QKVW (3 * ND)         // 1152
// (1/sqrt(384)) * log2(e): softmax computed in exp2 domain
#define ATT_SCALE2 0.07362221885f

// ---------------------------------------------------------------------------
// Embedding: x[b,t,:] = tok_emb[idx[b,t]] + pos_emb[t]
// ---------------------------------------------------------------------------
__global__ void __launch_bounds__(128) embed_kernel(const int* __restrict__ idx,
                                                    const float* __restrict__ tok,
                                                    const float* __restrict__ pos,
                                                    float* __restrict__ x) {
  int row = blockIdx.x;
  int t = row & (NT - 1);
  int token = idx[row];
  float* xr = x + (size_t)row * ND;
  const float* tr = tok + (size_t)token * ND;
  const float* pr = pos + (size_t)t * ND;
  for (int i = threadIdx.x; i < ND; i += 128) xr[i] = tr[i] + pr[i];
}

// ---------------------------------------------------------------------------
// LayerNorm (fp32 in -> f16 out), one row per 128-thread block.
// DPP 16-lane butterflies + 8 LDS partials: only two __syncthreads per row.
// ---------------------------------------------------------------------------
__global__ void __launch_bounds__(128) ln_kernel(const float* __restrict__ x,
                                                 const float* __restrict__ g,
                                                 const float* __restrict__ beta,
                                                 h16t* __restrict__ out) {
  __shared__ float part[16];
  int row = blockIdx.x, tid = threadIdx.x;
  int lane = tid & 31, wid = tid >> 5;
  const float* xr = x + (size_t)row * ND;
  float v0 = xr[tid], v1 = xr[tid + 128], v2 = xr[tid + 256];

  float s = red16_sum(v0 + v1 + v2);
  if ((lane & 15) == 0) part[wid * 2 + (lane >> 4)] = s;
  __syncthreads();
  float mean = 0.f;
#pragma unroll
  for (int i = 0; i < 8; ++i) mean += part[i];
  mean *= (1.f / (float)ND);

  float d0 = v0 - mean, d1 = v1 - mean, d2 = v2 - mean;
  float s2 = red16_sum(d0 * d0 + d1 * d1 + d2 * d2);
  if ((lane & 15) == 0) part[8 + wid * 2 + (lane >> 4)] = s2;
  __syncthreads();
  float var = 0.f;
#pragma unroll
  for (int i = 0; i < 8; ++i) var += part[8 + i];
  float rstd = rsqrtf(var * (1.f / (float)ND) + 1e-3f);

  h16t* orow = out + (size_t)row * ND;
  orow[tid]       = (h16t)(d0 * rstd * g[tid]       + beta[tid]);
  orow[tid + 128] = (h16t)(d1 * rstd * g[tid + 128] + beta[tid + 128]);
  orow[tid + 256] = (h16t)(d2 * rstd * g[tid + 256] + beta[tid + 256]);
}

// ---------------------------------------------------------------------------
// Weight conversion kernels (fp32 -> f16)
// ---------------------------------------------------------------------------
__global__ void __launch_bounds__(256) pack_qkv_kernel(const float* __restrict__ Wq,
                                                       const float* __restrict__ Wk,
                                                       const float* __restrict__ Wv,
                                                       h16t* __restrict__ dst,
                                                       int total) {
  int i = blockIdx.x * 256 + threadIdx.x;
  if (i >= total) return;
  int l = i / (ND * ND);
  int r = i % (ND * ND);
  int d = r / ND;
  int c = r % ND;                       // c = h*64 + hd
  int h = c >> 6, kk = c & 63;
  size_t src = (((size_t)l * NH + h) * ND + d) * NHD + kk;
  size_t dr = ((size_t)l * ND + d) * QKVW + c;
  dst[dr]            = (h16t)Wq[src];
  dst[dr + ND]       = (h16t)Wk[src];
  dst[dr + 2 * ND]   = (h16t)Wv[src];
}

__global__ void __launch_bounds__(256) conv_f16_kernel(const float* __restrict__ s,
                                                       h16t* __restrict__ d, int n) {
  int i = blockIdx.x * 256 + threadIdx.x;
  if (i < n) d[i] = (h16t)s[i];
}

// Whead [384,65] -> [384,128] zero-padded
__global__ void __launch_bounds__(256) conv_head_kernel(const float* __restrict__ s,
                                                        h16t* __restrict__ d) {
  int i = blockIdx.x * 256 + threadIdx.x;
  if (i >= ND * 128) return;
  int row = i >> 7, col = i & 127;
  d[i] = (col < NV) ? (h16t)s[row * NV + col] : (h16t)0.f;
}

// ---------------------------------------------------------------------------
// Generic WMMA GEMM: C[M,N] = A[M,K] @ B[K,N] (+bias, + epilogue mode)
// Block tile 128x128, BK=32, 8 waves: each wave owns a 32x64 region
// (2x4 C tiles of 16x16 -> 8 WMMA per K step per wave).
// Software pipelined: next K slab is fetched into registers while the
// current slab computes, so L2 latency overlaps the WMMA stream.
// ---------------------------------------------------------------------------
#define MODE_F16      0
#define MODE_F16_RELU 1
#define MODE_F32_ADD  2
#define MODE_F32_OUT  3

template <int MODE>
__global__ void __launch_bounds__(256) gemm_kernel(const h16t* __restrict__ A, int lda,
                                                   const h16t* __restrict__ B, int ldb,
                                                   int K, const float* __restrict__ bias,
                                                   void* __restrict__ outp, int ldo,
                                                   int Nreal) {
  __shared__ h16t Asm[128 * 40];   // A tile 128x32, stride 40 halves
  __shared__ h16t Bsm[128 * 40];   // B tile transposed: [n=128][k=32], stride 40

  int tid = threadIdx.x, lane = tid & 31, wid = tid >> 5;
  int wr = wid >> 1, wc = wid & 1;
  int m0 = blockIdx.x * 128, n0 = blockIdx.y * 128;

  v8f acc[2][4];
#pragma unroll
  for (int i = 0; i < 2; ++i)
#pragma unroll
    for (int j = 0; j < 4; ++j) acc[i][j] = zero8();

  uint4 aReg[2], bReg[2];
  auto load_tile = [&](int k0) {
#pragma unroll
    for (int v = 0; v < 2; ++v) {
      int li = v * 2048 + tid * 8;
      int ra = li >> 5, ca = li & 31;
      aReg[v] = *(const uint4*)(A + (size_t)(m0 + ra) * lda + k0 + ca);
      int rb = li >> 7, cb = li & 127;
      bReg[v] = *(const uint4*)(B + (size_t)(k0 + rb) * ldb + n0 + cb);
    }
  };
  load_tile(0);

  for (int k0 = 0; k0 < K; k0 += 32) {
    __syncthreads();                 // previous compute done reading LDS
    // commit staged registers to LDS
#pragma unroll
    for (int v = 0; v < 2; ++v) {
      int li = v * 2048 + tid * 8;
      int ra = li >> 5, ca = li & 31;
      *(uint4*)&Asm[ra * 40 + ca] = aReg[v];
      int rb = li >> 7, cb = li & 127;
      union { uint4 q; h16t h[8]; } d4;
      d4.q = bReg[v];
#pragma unroll
      for (int i = 0; i < 8; ++i) Bsm[(cb + i) * 40 + rb] = d4.h[i];
    }
    // issue next slab's global loads before the barrier: their latency
    // overlaps this slab's WMMAs
    if (k0 + 32 < K) load_tile(k0 + 32);
    __syncthreads();

    v16h a0 = load_a_frag(&Asm[(wr * 32) * 40], 40);
    v16h a1 = load_a_frag(&Asm[(wr * 32 + 16) * 40], 40);
    v16h bf[4];
#pragma unroll
    for (int j = 0; j < 4; ++j)
      bf[j] = load_b_frag(&Bsm[(wc * 64 + j * 16) * 40], 40);
#pragma unroll
    for (int j = 0; j < 4; ++j) {
      acc[0][j] = wmma32f16(a0, bf[j], acc[0][j]);
      acc[1][j] = wmma32f16(a1, bf[j], acc[1][j]);
    }
  }

  int rbase = m0 + wr * 32;
  int cbase = n0 + wc * 64;
  int radd = (lane & 16) ? 8 : 0;
#pragma unroll
  for (int i = 0; i < 2; ++i) {
#pragma unroll
    for (int j = 0; j < 4; ++j) {
      int col = cbase + j * 16 + (lane & 15);
      float bv = 0.f;
      if (MODE == MODE_F32_OUT) {
        if (bias != nullptr && col < Nreal) bv = bias[col];
      } else if (bias != nullptr) {
        bv = bias[col];
      }
#pragma unroll
      for (int r = 0; r < 8; ++r) {
        int row = rbase + i * 16 + r + radd;
        float cv = acc[i][j][r] + bv;
        if (MODE == MODE_F16) {
          ((h16t*)outp)[(size_t)row * ldo + col] = (h16t)cv;
        } else if (MODE == MODE_F16_RELU) {
          ((h16t*)outp)[(size_t)row * ldo + col] = (h16t)fmaxf(cv, 0.f);
        } else if (MODE == MODE_F32_ADD) {
          ((float*)outp)[(size_t)row * ldo + col] += cv;
        } else {
          if (col < Nreal) ((float*)outp)[(size_t)row * ldo + col] = cv;
        }
      }
    }
  }
}

// ---------------------------------------------------------------------------
// Fused causal attention per (b,h), FlashAttention-2 style:
//   pass 1: row max only (raw score domain; compare-select, no canonicalize,
//           no exponentials).
//   pass 2: ONE v_exp_f32 per element: e = exp2(fma(c, scale*log2e, -mrow));
//           accumulates the per-lane row sum in the same sweep, stages
//           unnormalized P for the P@V WMMAs, normalizes O at the store.
// Cross-lane reductions: DPP butterflies (VALU pipe), once per Q tile.
// qkv layout: [b*T+t][1152]: q at col h*64, k at 384+h*64, v at 768+h*64.
// ---------------------------------------------------------------------------
__global__ void __launch_bounds__(256) attn_kernel(const h16t* __restrict__ qkv,
                                                   h16t* __restrict__ attC) {
  __shared__ h16t vT[64 * 264];          // V transposed: [d][t], stride 264
  __shared__ h16t Pst[8 * 16 * 40];      // per-wave P staging tile 16x32

  int tid = threadIdx.x, lane = tid & 31, wid = tid >> 5;
  int b = blockIdx.x / NH, h = blockIdx.x % NH;

  // load V^T
  {
    int t = tid;
    const h16t* vs = qkv + ((size_t)(b * NT + t)) * QKVW + 2 * ND + h * NHD;
#pragma unroll
    for (int d = 0; d < NHD; ++d) vT[d * 264 + t] = vs[d];
  }
  __syncthreads();

  const h16t* kbase = qkv + ((size_t)(b * NT)) * QKVW + ND + h * NHD;
  h16t* Ps = &Pst[wid * 16 * 40];

  for (int qq = 0; qq < 2; ++qq) {
    int qt = wid + qq * 8;               // q row-tile index 0..15
    int r0 = qt * 16;
    const h16t* qbase = qkv + ((size_t)(b * NT + r0)) * QKVW + h * NHD;
    v16h aq0 = load_a_frag(qbase, QKVW);
    v16h aq1 = load_a_frag(qbase + 32, QKVW);
    int radd = (lane & 16) ? 8 : 0;

    // ---- pass 1: per-lane row max in RAW score domain (cmp-select only) ----
    float mlan[8];
#pragma unroll
    for (int r = 0; r < 8; ++r) mlan[r] = -1e30f;

    for (int j = 0; j <= qt; ++j) {      // causal: only k-tiles <= qt
      v16h bk0 = load_b_frag(kbase + (size_t)(j * 16) * QKVW, QKVW);
      v16h bk1 = load_b_frag(kbase + (size_t)(j * 16) * QKVW + 32, QKVW);
      v8f c = zero8();
      c = wmma32f16(aq0, bk0, c);
      c = wmma32f16(aq1, bk1, c);
      int col = j * 16 + (lane & 15);
#pragma unroll
      for (int r = 0; r < 8; ++r) {
        int row = r0 + r + radd;
        float sv = (col > row) ? -1e30f : c[r];
        mlan[r] = (sv > mlan[r]) ? sv : mlan[r];
      }
    }
    // merge max across the 16-lane row group, then scale once
    float mrow[8];
#pragma unroll
    for (int r = 0; r < 8; ++r) mrow[r] = red16_max(mlan[r]) * ATT_SCALE2;

    // ---- pass 2: e = exp2(c*scale2 - mrow), accumulate sum, O += P@V ----
    float slan[8];
#pragma unroll
    for (int r = 0; r < 8; ++r) slan[r] = 0.f;
    v8f oacc[4];
#pragma unroll
    for (int nt = 0; nt < 4; ++nt) oacc[nt] = zero8();

    for (int j2 = 0; j2 <= qt; j2 += 2) {   // pairs of k-tiles -> K=32 for P@V
#pragma unroll
      for (int jj = 0; jj < 2; ++jj) {
        int j = j2 + jj;
        if (j <= qt) {                      // uniform per wave
          v16h bk0 = load_b_frag(kbase + (size_t)(j * 16) * QKVW, QKVW);
          v16h bk1 = load_b_frag(kbase + (size_t)(j * 16) * QKVW + 32, QKVW);
          v8f c = zero8();
          c = wmma32f16(aq0, bk0, c);
          c = wmma32f16(aq1, bk1, c);
          int col = j * 16 + (lane & 15);
#pragma unroll
          for (int r = 0; r < 8; ++r) {
            int row = r0 + r + radd;
            float arg = fmaf(c[r], ATT_SCALE2, -mrow[r]);
            arg = (col > row) ? -1e30f : arg;
            float e = exp2raw(arg);         // masked -> exp2(-1e30) = 0
            slan[r] += e;
            Ps[(r + radd) * 40 + jj * 16 + (lane & 15)] = (h16t)e;
          }
        } else {
#pragma unroll
          for (int r = 0; r < 8; ++r)
            Ps[(r + radd) * 40 + jj * 16 + (lane & 15)] = (h16t)0.f;
        }
      }
      v16h ap = load_a_frag(Ps, 40);        // unnormalized P tile 16x32
#pragma unroll
      for (int nt = 0; nt < 4; ++nt) {
        v16h bv = load_b_frag(&vT[(nt * 16) * 264 + j2 * 16], 264);
        oacc[nt] = wmma32f16(ap, bv, oacc[nt]);
      }
    }

    // merge row sums across lanes, normalize O at store
    float inv[8];
#pragma unroll
    for (int r = 0; r < 8; ++r) inv[r] = 1.f / red16_sum(slan[r]);

#pragma unroll
    for (int nt = 0; nt < 4; ++nt) {
      int d = nt * 16 + (lane & 15);
#pragma unroll
      for (int r = 0; r < 8; ++r) {
        int row = r0 + r + radd;
        attC[((size_t)(b * NT + row)) * ND + h * NHD + d] =
            (h16t)(oacc[nt][r] * inv[r]);
      }
    }
  }
}

// ---------------------------------------------------------------------------
// Host orchestration
// ---------------------------------------------------------------------------
extern "C" void kernel_launch(void* const* d_in, const int* in_sizes, int n_in,
                              void* d_out, int out_size, void* d_ws, size_t ws_size,
                              hipStream_t stream) {
  (void)in_sizes; (void)n_in; (void)out_size; (void)ws_size;

  const int*   idx   = (const int*)  d_in[0];
  const float* tok   = (const float*)d_in[1];
  const float* pos   = (const float*)d_in[2];
  const float* Wq    = (const float*)d_in[3];
  const float* Wk    = (const float*)d_in[4];
  const float* Wv    = (const float*)d_in[5];
  const float* Wproj = (const float*)d_in[6];
  const float* bproj = (const float*)d_in[7];
  const float* W1    = (const float*)d_in[8];
  const float* b1    = (const float*)d_in[9];
  const float* W2    = (const float*)d_in[10];
  const float* b2    = (const float*)d_in[11];
  const float* ln1g  = (const float*)d_in[12];
  const float* ln1b  = (const float*)d_in[13];
  const float* ln2g  = (const float*)d_in[14];
  const float* ln2b  = (const float*)d_in[15];
  const float* lnfg  = (const float*)d_in[16];
  const float* lnfb  = (const float*)d_in[17];
  const float* Whead = (const float*)d_in[18];
  const float* bhead = (const float*)d_in[19];

  // workspace layout (bytes)
  char* ws = (char*)d_ws;
  float* x       = (float*)(ws + 0);              // 16384*384 f32    25.2 MB
  h16t*  hbuf    = (h16t*)(ws + 25165824);        // 16384*384 f16
  h16t*  qkv16   = (h16t*)(ws + 37748736);        // 16384*1152 f16
  h16t*  attC    = (h16t*)(ws + 75497472);        // 16384*384 f16
  h16t*  m16     = (h16t*)(ws + 88080384);        // 16384*1536 f16
  h16t*  Wqkv16  = (h16t*)(ws + 138412032);       // 6*384*1152 f16
  h16t*  Wproj16 = (h16t*)(ws + 143720448);       // 6*384*384 f16
  h16t*  W1_16   = (h16t*)(ws + 145489920);       // 6*384*1536 f16
  h16t*  W2_16   = (h16t*)(ws + 152567808);       // 6*1536*384 f16
  h16t*  Wh16    = (h16t*)(ws + 159645696);       // 384*128 f16 (padded)

  // weight conversion
  pack_qkv_kernel<<<(884736 + 255) / 256, 256, 0, stream>>>(Wq, Wk, Wv, Wqkv16, 884736);
  conv_f16_kernel<<<(884736 + 255) / 256, 256, 0, stream>>>(Wproj, Wproj16, 884736);
  conv_f16_kernel<<<(3538944 + 255) / 256, 256, 0, stream>>>(W1, W1_16, 3538944);
  conv_f16_kernel<<<(3538944 + 255) / 256, 256, 0, stream>>>(W2, W2_16, 3538944);
  conv_head_kernel<<<(ND * 128 + 255) / 256, 256, 0, stream>>>(Whead, Wh16);

  embed_kernel<<<BT, 128, 0, stream>>>(idx, tok, pos, x);

  for (int l = 0; l < NL; ++l) {
    ln_kernel<<<BT, 128, 0, stream>>>(x, ln1g + l * ND, ln1b + l * ND, hbuf);
    gemm_kernel<MODE_F16><<<dim3(BT / 128, QKVW / 128), 256, 0, stream>>>(
        hbuf, ND, Wqkv16 + (size_t)l * ND * QKVW, QKVW, ND, nullptr, qkv16, QKVW, 0);
    attn_kernel<<<NB * NH, 256, 0, stream>>>(qkv16, attC);
    gemm_kernel<MODE_F32_ADD><<<dim3(BT / 128, ND / 128), 256, 0, stream>>>(
        attC, ND, Wproj16 + (size_t)l * ND * ND, ND, ND, bproj + l * ND, x, ND, 0);
    ln_kernel<<<BT, 128, 0, stream>>>(x, ln2g + l * ND, ln2b + l * ND, hbuf);
    gemm_kernel<MODE_F16_RELU><<<dim3(BT / 128, DFF / 128), 256, 0, stream>>>(
        hbuf, ND, W1_16 + (size_t)l * ND * DFF, DFF, ND, b1 + l * DFF, m16, DFF, 0);
    gemm_kernel<MODE_F32_ADD><<<dim3(BT / 128, ND / 128), 256, 0, stream>>>(
        m16, DFF, W2_16 + (size_t)l * DFF * ND, ND, DFF, b2 + l * ND, x, ND, 0);
  }

  ln_kernel<<<BT, 128, 0, stream>>>(x, lnfg, lnfb, hbuf);
  gemm_kernel<MODE_F32_OUT><<<dim3(BT / 128, 1), 256, 0, stream>>>(
      hbuf, ND, Wh16, 128, ND, bhead, (float*)d_out, NV, NV);
}
